// DynamicGraphRiskModel_27608049778854
// MI455X (gfx1250) — matrix operand order repaired and checked by reading, not compiled, not run on hardware
//
#include <hip/hip_runtime.h>

#define D 128
#define XS_STRIDE 40   // 32 k + 8 pad (bf16 elems) -> 80B rows, keeps 16B alignment

typedef __attribute__((ext_vector_type(16))) __bf16 v16bf;
typedef __attribute__((ext_vector_type(8)))  float  v8f;

union Frag16 {
  v16bf v;
  uint4 q[2];
  unsigned short u[16];
};

__device__ __forceinline__ unsigned short f2bf(float f) {
  unsigned int u = __float_as_uint(f);
  unsigned int r = (u + 0x7FFFu + ((u >> 16) & 1u)) >> 16;  // round-nearest-even
  return (unsigned short)r;
}

__device__ __forceinline__ unsigned pack2(unsigned short lo, unsigned short hi) {
  return (unsigned)lo | ((unsigned)hi << 16);
}

__global__ void zero_f32_kernel(float* __restrict__ p, int n) {
  int i = blockIdx.x * blockDim.x + threadIdx.x;
  if (i < n) p[i] = 0.0f;
}

__global__ void degree_kernel(const int* __restrict__ dst, float* __restrict__ deg, int e) {
  int i = blockIdx.x * blockDim.x + threadIdx.x;
  if (i < e) atomicAdd(&deg[dst[i]], 1.0f);
}

__global__ void invdeg_kernel(float* __restrict__ deg, int n) {
  int i = blockIdx.x * blockDim.x + threadIdx.x;
  if (i < n) deg[i] = 1.0f / fmaxf(deg[i], 1.0f);
}

// one thread per (edge, 4-float chunk): float4 gather + 4 f32 atomic adds (L2-resident)
__global__ void scatter_kernel(const float* __restrict__ x, const int* __restrict__ src,
                               const int* __restrict__ dst, float* __restrict__ agg, int e) {
  int t = blockIdx.x * blockDim.x + threadIdx.x;
  const int chunks = D / 4;  // 32
  if (t >= e * chunks) return;
  int edge = t / chunks;
  int c    = (t % chunks) * 4;
  int s  = src[edge];
  int d2 = dst[edge];
  const float4 val = *(const float4*)&x[(long)s * D + c];
  float* base = &agg[(long)d2 * D + c];
  atomicAdd(base + 0, val.x);
  atomicAdd(base + 1, val.y);
  atomicAdd(base + 2, val.z);
  atomicAdd(base + 3, val.w);
}

// out = act( X @ Ws  +  (AGG * invdeg) @ Wn  + bias ), K = ksteps*32 (self k 0..127, neigh 128..255)
// block: 256 threads = 8 waves; block tile 128 rows x 128 cols; wave tile 16 x 128.
// Double-buffered LDS staging; bf16 WMMA 16x16x32 with f32 accumulation.
__global__ __launch_bounds__(256)
void sage_gemm_kernel(const float* __restrict__ X, const float* __restrict__ AGG,
                      const float* __restrict__ invdeg,
                      const float* __restrict__ Ws, const float* __restrict__ Wn,
                      const float* __restrict__ bias, float* __restrict__ out,
                      int nrows, int ksteps, int do_relu) {
  __shared__ unsigned short Xs[2][128 * XS_STRIDE];  // 2 x 10 KB: bf16 A tiles 128x32
  __shared__ unsigned short Wp[2][8 * 32 * 16];      // 2 x  8 KB: bf16 B fragments, per-lane order

  const int tid  = threadIdx.x;
  const int lane = tid & 31;
  const int wid  = tid >> 5;
  const int row0 = blockIdx.x * 128;

  // Stage k-step `ks` into LDS buffer `buf`. Self/neigh selection is uniform per k-step
  // (kbase is a multiple of 32), so hoist it; row guard becomes an index clamp.
  auto stage = [&](int ks, int buf) {
    const int kbase   = ks * 32;
    const bool neigh  = (kbase >= D);
    const float* __restrict__ srcX = neigh ? AGG : X;
    const float* __restrict__ srcW = neigh ? Wn : Ws;
    const int kb = neigh ? (kbase - D) : kbase;

    // A tile: each chunk = 8 consecutive k of one row -> one ds_store_b128
    for (int c = tid; c < 128 * 4; c += 256) {
      int r  = c >> 2;
      int kc = (c & 3) * 8;
      int g  = row0 + r;
      if (g >= nrows) g = nrows - 1;                 // clamp: padded rows are don't-care
      const float* p = &srcX[(long)g * D + kb + kc];
      float sc = neigh ? invdeg[g] : 1.0f;
      float4 a = *(const float4*)p;
      float4 b = *(const float4*)(p + 4);
      uint4 o;
      o.x = pack2(f2bf(a.x * sc), f2bf(a.y * sc));
      o.y = pack2(f2bf(a.z * sc), f2bf(a.w * sc));
      o.z = pack2(f2bf(b.x * sc), f2bf(b.y * sc));
      o.w = pack2(f2bf(b.z * sc), f2bf(b.w * sc));
      *(uint4*)&Xs[buf][r * XS_STRIDE + kc] = o;
    }

    // B fragments packed in the ISA per-lane layout:
    //   lanes 0-15:  N = lane,    K = kb + 0..7, 16..23 (pairs per VGPR)
    //   lanes 16-31: N = lane-16, K = kb + 8..15, 24..31
    for (int c = tid; c < 8 * 32 * 2; c += 256) {
      int ct   = c >> 6;          // column tile 0..7
      int ln   = (c >> 1) & 31;   // lane
      int half = c & 1;           // low/high 8 elements
      int col  = ct * 16 + (ln & 15);
      int k0   = kb + ((ln < 16) ? 0 : 8) + half * 16;
      unsigned short t[8];
#pragma unroll
      for (int j = 0; j < 8; ++j) t[j] = f2bf(srcW[(k0 + j) * D + col]);
      uint4 o;
      o.x = pack2(t[0], t[1]);
      o.y = pack2(t[2], t[3]);
      o.z = pack2(t[4], t[5]);
      o.w = pack2(t[6], t[7]);
      *(uint4*)&Wp[buf][ct * 512 + ln * 16 + half * 8] = o;
    }
  };

  v8f acc[8];
  {
    v8f z = {};
    for (int ct = 0; ct < 8; ++ct) acc[ct] = z;
  }

  const int rowl = wid * 16 + (lane & 15);
  const int aoff = (lane < 16) ? 0 : 8;

  stage(0, 0);
  __syncthreads();

  for (int ks = 0; ks < ksteps; ++ks) {
    const int buf = ks & 1;
    if (ks + 1 < ksteps) stage(ks + 1, buf ^ 1);  // overlap next-tile staging with WMMA

    // A fragment (16x32 bf16), ISA layout:
    //   lanes 0-15:  M=lane,    u[0..7]=K 0..7,  u[8..15]=K 16..23
    //   lanes 16-31: M=lane-16, u[0..7]=K 8..15, u[8..15]=K 24..31
    Frag16 afr;
    afr.q[0] = *(const uint4*)&Xs[buf][rowl * XS_STRIDE + aoff];
    afr.q[1] = *(const uint4*)&Xs[buf][rowl * XS_STRIDE + aoff + 16];

#pragma unroll
    for (int ct = 0; ct < 8; ++ct) {
      Frag16 bfr;
      bfr.q[0] = *(const uint4*)&Wp[buf][ct * 512 + lane * 16];
      bfr.q[1] = *(const uint4*)&Wp[buf][ct * 512 + lane * 16 + 8];
      acc[ct] = __builtin_amdgcn_wmma_f32_16x16x32_bf16(
          false, afr.v, false, bfr.v, (short)0, acc[ct], false, false);
    }
    __syncthreads();
  }

  // epilogue: C layout (VGPR r: lanes 0-15 -> M=r, N=lane; lanes 16-31 -> M=8+r, N=lane-16)
  const int colb = lane & 15;
  const int moff = (lane < 16) ? 0 : 8;
  for (int ct = 0; ct < 8; ++ct) {
    int col = ct * 16 + colb;
    float bv = bias[col];
#pragma unroll
    for (int r = 0; r < 8; ++r) {
      int g = row0 + wid * 16 + moff + r;
      if (g < nrows) {
        float v = acc[ct][r] + bv;
        if (do_relu) v = fmaxf(v, 0.0f);
        out[(long)g * D + col] = v;
      }
    }
  }
}

// out[n, o] = bc2[o] + sum_k hc[n,k] * Wc2[k,o], OUT=2
__global__ void classifier_out_kernel(const float* __restrict__ hc, const float* __restrict__ Wc2,
                                      const float* __restrict__ bc2, float* __restrict__ out, int n) {
  int t = blockIdx.x * blockDim.x + threadIdx.x;
  if (t >= n * 2) return;
  int node = t >> 1;
  int o    = t & 1;
  float s = bc2[o];
  const float* row = &hc[(long)node * D];
#pragma unroll 4
  for (int k = 0; k < D; ++k) s += row[k] * Wc2[k * 2 + o];
  out[t] = s;
}

extern "C" void kernel_launch(void* const* d_in, const int* in_sizes, int n_in,
                              void* d_out, int out_size, void* d_ws, size_t ws_size,
                              hipStream_t stream) {
  (void)n_in; (void)out_size; (void)ws_size;

  const float* x   = (const float*)d_in[0];
  const int*   src = (const int*)d_in[1];
  const int*   dst = (const int*)d_in[2];
  const float* Ws1 = (const float*)d_in[3];
  const float* Wn1 = (const float*)d_in[4];
  const float* b1  = (const float*)d_in[5];
  const float* Ws2 = (const float*)d_in[6];
  const float* Wn2 = (const float*)d_in[7];
  const float* b2  = (const float*)d_in[8];
  const float* Wc1 = (const float*)d_in[9];
  const float* bc1 = (const float*)d_in[10];
  const float* Wc2 = (const float*)d_in[11];
  const float* bc2 = (const float*)d_in[12];
  float* out = (float*)d_out;

  const int N = in_sizes[0] / D;   // 100000
  const int E = in_sizes[1];       // 1600000

  float* ws  = (float*)d_ws;
  float* deg = ws;                              // N
  float* agg = deg + N;                         // N*D (also reused as classifier hidden)
  float* h1  = agg + (size_t)N * D;             // N*D
  float* h2  = h1  + (size_t)N * D;             // N*D

  const dim3 blk(256);
  auto nblk = [](long t) { return dim3((unsigned)((t + 255) / 256)); };
  const int nd = N * D;
  const int gemm_blocks = (N + 127) / 128;
  const long scatter_threads = (long)E * (D / 4);

  // degrees (shared by both layers)
  zero_f32_kernel<<<nblk(N), blk, 0, stream>>>(deg, N);
  degree_kernel<<<nblk(E), blk, 0, stream>>>(dst, deg, E);
  invdeg_kernel<<<nblk(N), blk, 0, stream>>>(deg, N);

  // layer 1: h1 = relu(x@Ws1 + mean@Wn1 + b1)
  zero_f32_kernel<<<nblk(nd), blk, 0, stream>>>(agg, nd);
  scatter_kernel<<<nblk(scatter_threads), blk, 0, stream>>>(x, src, dst, agg, E);
  sage_gemm_kernel<<<gemm_blocks, blk, 0, stream>>>(x, agg, deg, Ws1, Wn1, b1, h1, N, 8, 1);

  // layer 2: h2 = h1@Ws2 + mean@Wn2 + b2   (no relu)
  zero_f32_kernel<<<nblk(nd), blk, 0, stream>>>(agg, nd);
  scatter_kernel<<<nblk(scatter_threads), blk, 0, stream>>>(h1, src, dst, agg, E);
  sage_gemm_kernel<<<gemm_blocks, blk, 0, stream>>>(h1, agg, deg, Ws2, Wn2, b2, h2, N, 8, 0);

  // classifier hidden: hc = relu(h2@Wc1 + bc1), self-path only (ksteps=4 => AGG never read);
  // result stored into agg (no longer needed)
  sage_gemm_kernel<<<gemm_blocks, blk, 0, stream>>>(h2, agg, deg, Wc1, Wc1, bc1, agg, N, 4, 1);

  // final: out = hc @ Wc2 + bc2
  classifier_out_kernel<<<nblk((long)N * 2), blk, 0, stream>>>(agg, Wc2, bc2, out, N);
}
